// EncoderAndMemory_9045201125671
// MI455X (gfx1250) — compile-verified, matrix operand order checked by reading, not run on hardware
//
#include <hip/hip_runtime.h>
#include <hip/hip_bf16.h>
#include <cstdint>

#define B_DIM   8192
#define LAT     256
#define MEM_DIM 200
#define IMG2    16384   // 128*128

typedef __bf16 v16bf __attribute__((ext_vector_type(16)));
typedef float  v8f   __attribute__((ext_vector_type(8)));

// ---- helpers ---------------------------------------------------------------

__device__ __forceinline__ unsigned short f2bf(float f) {
  // round-to-nearest-even f32 -> bf16 bits
  unsigned u = __builtin_bit_cast(unsigned, f);
  u += 0x7FFFu + ((u >> 16) & 1u);
  return (unsigned short)(u >> 16);
}

__device__ __forceinline__ float wave_sum(float v) {
#pragma unroll
  for (int o = 16; o; o >>= 1) v += __shfl_xor(v, o, 32);
  return v;
}

__device__ __forceinline__ float wave_max(float v) {
#pragma unroll
  for (int o = 16; o; o >>= 1) v = fmaxf(v, __shfl_xor(v, o, 32));
  return v;
}

// gfx1250 async global->LDS copy (16B per lane), tracked by ASYNCcnt.
__device__ __forceinline__ void async_b128(unsigned lds_off, const void* gptr) {
  asm volatile("global_load_async_to_lds_b128 %0, %1, off"
               :: "v"(lds_off), "v"(gptr)
               : "memory");
}

__device__ __forceinline__ void wait_async0() {
  asm volatile("s_wait_asynccnt 0x0" ::: "memory");
}

// ---- kernel 1: memory row L2 norms (200 rows, one wave per row) ------------

__global__ void k_memnorm(const float* __restrict__ memory,
                          float* __restrict__ mn) {
  const int m = blockIdx.x;
  const int l = threadIdx.x;          // 0..31
  const float* row = memory + (size_t)m * LAT;
  float s = 0.f;
#pragma unroll
  for (int j = l; j < LAT; j += 32) { float v = row[j]; s += v * v; }
  s = wave_sum(s);
  if (l == 0) mn[m] = sqrtf(s);
}

// ---- kernel 2: W f32 -> bf16 (vectorized, exact coverage) ------------------

__global__ void k_wconv(const float* __restrict__ W,
                        unsigned short* __restrict__ Wbf) {
  const size_t i4 = (size_t)blockIdx.x * blockDim.x + threadIdx.x; // float4 idx
  const float4 v = ((const float4*)W)[i4];
  union { unsigned short s[4]; uint2 u; } p;
  p.s[0] = f2bf(v.x); p.s[1] = f2bf(v.y);
  p.s[2] = f2bf(v.z); p.s[3] = f2bf(v.w);
  ((uint2*)Wbf)[i4] = p.u;
}

// ---- kernel 3: fused cosine-softmax-shrink-L1 + z_hat ----------------------
// One batch row per wave; 8 waves (256 threads) per block; 1024 blocks.

__global__ __launch_bounds__(256)
void k_attn(const float* __restrict__ z, const float* __restrict__ memory,
            const float* __restrict__ mn, float* __restrict__ w_out,
            float* __restrict__ zhat_out, unsigned short* __restrict__ zhat_bf) {
  __shared__ float lg[8][MEM_DIM];
  const int lane = threadIdx.x & 31;
  const int wid  = threadIdx.x >> 5;
  const int row  = blockIdx.x * 8 + wid;

  const float4* zp = (const float4*)(z + (size_t)row * LAT + lane * 8);
  const float4 z0 = zp[0], z1 = zp[1];

  float nn = z0.x*z0.x + z0.y*z0.y + z0.z*z0.z + z0.w*z0.w
           + z1.x*z1.x + z1.y*z1.y + z1.z*z1.z + z1.w*z1.w;
  const float zn = sqrtf(wave_sum(nn));

  for (int m = 0; m < MEM_DIM; ++m) {
    const float4* mp = (const float4*)(memory + (size_t)m * LAT + lane * 8);
    const float4 a = mp[0], b = mp[1];
    float d = z0.x*a.x + z0.y*a.y + z0.z*a.z + z0.w*a.w
            + z1.x*b.x + z1.y*b.y + z1.z*b.z + z1.w*b.w;
    d = wave_sum(d);
    if (lane == 0) lg[wid][m] = d / fmaxf(zn * mn[m], 1e-8f);
  }

  float mx = -3.4e38f;
  for (int m = lane; m < MEM_DIM; m += 32) mx = fmaxf(mx, lg[wid][m]);
  mx = wave_max(mx);

  float se = 0.f;
  for (int m = lane; m < MEM_DIM; m += 32) {
    const float e = __expf(lg[wid][m] - mx);
    lg[wid][m] = e; se += e;
  }
  se = wave_sum(se);

  const float T = 1.0f / (float)MEM_DIM;
  float l1 = 0.f;
  for (int m = lane; m < MEM_DIM; m += 32) {
    const float wv = lg[wid][m] / se;
    const float d  = wv - T;
    const float ws = fmaxf(d, 0.f) * wv / (fabsf(d) + 1e-12f);
    lg[wid][m] = ws; l1 += ws;
  }
  l1 = wave_sum(l1);
  const float inv = 1.0f / fmaxf(l1, 1e-12f);

  for (int m = lane; m < MEM_DIM; m += 32) {
    const float wf = lg[wid][m] * inv;
    lg[wid][m] = wf;
    w_out[(size_t)row * MEM_DIM + m] = wf;
  }

  float az[8] = {};
  for (int m = 0; m < MEM_DIM; ++m) {
    const float wm = lg[wid][m];
    const float4* mp = (const float4*)(memory + (size_t)m * LAT + lane * 8);
    const float4 a = mp[0], b = mp[1];
    az[0] += wm * a.x; az[1] += wm * a.y; az[2] += wm * a.z; az[3] += wm * a.w;
    az[4] += wm * b.x; az[5] += wm * b.y; az[6] += wm * b.z; az[7] += wm * b.w;
  }
  float4 o0 = {az[0], az[1], az[2], az[3]};
  float4 o1 = {az[4], az[5], az[6], az[7]};
  float4* op = (float4*)(zhat_out + (size_t)row * LAT + lane * 8);
  op[0] = o0; op[1] = o1;

  union { unsigned short s[8]; uint4 u; } pk;
#pragma unroll
  for (int j = 0; j < 8; ++j) pk.s[j] = f2bf(az[j]);
  *((uint4*)(zhat_bf + (size_t)row * LAT + lane * 8)) = pk.u;
}

// ---- kernel 4: x_hat = z_hat(bf16) @ W(bf16)^T + b via WMMA ----------------
// Block tile 128(M) x 128(N), 256 threads = 8 waves in 4(M) x 2(N).
// Wave tile 32x64 -> 2x4 accumulators; K = 256 = 8 slabs of 32.
// Each K-slab of A (128x32) and B (128x32) is staged once into LDS via
// async global->LDS b128 copies (double buffered), so the next slab's DMA
// overlaps this slab's ds_load_b128 + v_wmma work. Unique L2 reads: 128 KB
// per 64 KB output tile (~1 GB total), well under the 23 us HBM store floor.

__device__ __forceinline__ void stage_fill(const unsigned short* __restrict__ Abf,
                                           const unsigned short* __restrict__ Wbf,
                                           int rowM0, int colN0, int kk,
                                           unsigned ldsA, unsigned ldsB, int t) {
#pragma unroll
  for (int i = 0; i < 2; ++i) {
    const int cid = t + i * 256;        // [0,512): 128 rows x 4 16B chunks
    const int r = cid >> 2, p = cid & 3;
    async_b128(ldsA + r * 64 + p * 16,
               Abf + (size_t)(rowM0 + r) * LAT + kk * 32 + p * 8);
    async_b128(ldsB + r * 64 + p * 16,
               Wbf + (size_t)(colN0 + r) * LAT + kk * 32 + p * 8);
  }
}

__global__ __launch_bounds__(256)
void k_gemm(const unsigned short* __restrict__ Abf,   // [8192,256] bf16
            const unsigned short* __restrict__ Wbf,   // [16384,256] bf16
            const float* __restrict__ bias,           // [16384]
            float* __restrict__ out) {                // [8192,16384]
  __shared__ unsigned short shA[2][128][32];          // 16 KB
  __shared__ unsigned short shB[2][128][32];          // 16 KB

  const int t     = threadIdx.x;
  const int lane  = t & 31;
  const int wid   = t >> 5;
  const int waveM = wid >> 1;                 // 0..3
  const int waveN = wid & 1;                  // 0..1
  const int lm    = lane & 15;
  const int h     = lane >> 4;

  const int rowM0 = blockIdx.y * 128;
  const int colN0 = blockIdx.x * 128;

  const unsigned aBase = (unsigned)(uintptr_t)&shA[0][0][0];
  const unsigned bBase = (unsigned)(uintptr_t)&shB[0][0][0];

  v8f acc[2][4] = {};

  // prologue: stage slab 0 into buffer 0
  stage_fill(Abf, Wbf, rowM0, colN0, 0, aBase, bBase, t);
  wait_async0();
  __syncthreads();

#pragma unroll
  for (int kk = 0; kk < 8; ++kk) {
    const int buf = kk & 1;
    if (kk < 7)
      stage_fill(Abf, Wbf, rowM0, colN0, kk + 1,
                 aBase + (unsigned)((buf ^ 1) * 8192),
                 bBase + (unsigned)((buf ^ 1) * 8192), t);

    union Frag { uint4 u[2]; v16bf v; };
    Frag a[2], b[4];
#pragma unroll
    for (int af = 0; af < 2; ++af) {
      const int r16 = waveM * 32 + af * 16 + lm;
      a[af].u[0] = *(const uint4*)&shA[buf][r16][h * 8];
      a[af].u[1] = *(const uint4*)&shA[buf][r16][16 + h * 8];
    }
#pragma unroll
    for (int bf = 0; bf < 4; ++bf) {
      const int c16 = waveN * 64 + bf * 16 + lm;
      b[bf].u[0] = *(const uint4*)&shB[buf][c16][h * 8];
      b[bf].u[1] = *(const uint4*)&shB[buf][c16][16 + h * 8];
    }
#pragma unroll
    for (int af = 0; af < 2; ++af)
#pragma unroll
      for (int bf = 0; bf < 4; ++bf)
        acc[af][bf] = __builtin_amdgcn_wmma_f32_16x16x32_bf16(
            /*neg_a=*/false, a[af].v, /*neg_b=*/false, b[bf].v,
            /*c_mod=*/(short)0, acc[af][bf],
            /*reuse_a=*/false, /*reuse_b=*/false);

    if (kk < 7) { wait_async0(); __syncthreads(); }
  }

  // D layout: VGPR r -> (row = base + h*8 + r, col = base + lm); bias fused,
  // non-temporal stores keep the 536 MB x_hat stream out of L2.
#pragma unroll
  for (int af = 0; af < 2; ++af) {
#pragma unroll
    for (int bf = 0; bf < 4; ++bf) {
      const int col = colN0 + waveN * 64 + bf * 16 + lm;
      const float bb = bias[col];
#pragma unroll
      for (int r = 0; r < 8; ++r) {
        const int row = rowM0 + waveM * 32 + af * 16 + h * 8 + r;
        __builtin_nontemporal_store(acc[af][bf][r] + bb,
                                    &out[(size_t)row * IMG2 + col]);
      }
    }
  }
}

// ---- launcher --------------------------------------------------------------

extern "C" void kernel_launch(void* const* d_in, const int* in_sizes, int n_in,
                              void* d_out, int out_size, void* d_ws, size_t ws_size,
                              hipStream_t stream) {
  const float* z      = (const float*)d_in[0];   // [8192,256]
  const float* memory = (const float*)d_in[1];   // [200,256]
  const float* W      = (const float*)d_in[2];   // [16384,256]
  const float* bias   = (const float*)d_in[3];   // [16384]

  // d_out: x_hat [8192,16384] ++ z_hat [8192,256] ++ w [8192,200]
  float* xhat = (float*)d_out;
  float* zhat = xhat + (size_t)B_DIM * IMG2;
  float* wout = zhat + (size_t)B_DIM * LAT;

  // workspace: z_hat bf16 (4 MB) | W bf16 (8 MB) | mn (800 B)
  char* wsb = (char*)d_ws;
  unsigned short* zhat_bf = (unsigned short*)wsb;
  unsigned short* Wbf     = (unsigned short*)(wsb + (size_t)B_DIM * LAT * 2);
  float* mn = (float*)(wsb + (size_t)B_DIM * LAT * 2 + (size_t)IMG2 * LAT * 2);

  k_memnorm<<<MEM_DIM, 32, 0, stream>>>(memory, mn);
  k_wconv<<<(IMG2 * LAT / 4) / 256, 256, 0, stream>>>(W, Wbf);
  k_attn<<<B_DIM / 8, 256, 0, stream>>>(z, memory, mn, wout, zhat, zhat_bf);

  dim3 g2(IMG2 / 128, B_DIM / 128);   // (128, 64) blocks of 128x128 output
  k_gemm<<<g2, 256, 0, stream>>>(zhat_bf, Wbf, bias, xhat);
}
